// EdgeBlock_2740189135079
// MI455X (gfx1250) — matrix operand order repaired and checked by reading, not compiled, not run on hardware
//
#include <hip/hip_runtime.h>

// EdgeBlock: out[b,e,:] = relu(concat(edata, vdata[send], vdata[recv], cdata) @ W + b)
// Implemented as an f16-WMMA GEMM (f32 accumulate) with LDS-staged, gathered A tiles.

typedef _Float16 v16h __attribute__((ext_vector_type(16)));
typedef _Float16 v8h  __attribute__((ext_vector_type(8)));
typedef _Float16 v4h  __attribute__((ext_vector_type(4)));
typedef float    v8f  __attribute__((ext_vector_type(8)));
typedef float    v4f  __attribute__((ext_vector_type(4)));

#define E_PER_B  400000
#define NVERT    50000
#define DE       64
#define DV       64
#define DC       32
#define DIN      224
#define DOUT     64
#define LDK      232        // padded K stride (halves) for LDS tiles
#define ROWS     64         // edges per super-tile (4 waves x 16)
#define NTHREADS 128

__global__ __launch_bounds__(NTHREADS)
void edgeblock_wmma(const float* __restrict__ edata,
                    const float* __restrict__ vdata,
                    const float* __restrict__ cdata,
                    const int*   __restrict__ sender,
                    const int*   __restrict__ receiver,
                    const float* __restrict__ W,
                    const float* __restrict__ bias,
                    float* __restrict__ out,
                    int total, int n_super)
{
    __shared__ _Float16 sW[DOUT * LDK];    // W transposed: sW[o*LDK + k], f16
    __shared__ _Float16 sAct[ROWS * LDK];  // activations:  sAct[row*LDK + k], f16
    __shared__ float    sBias[DOUT];

    const int tid  = (int)threadIdx.x;
    const int lane = tid & 31;
    const int wave = tid >> 5;

    // ---- one-time per block: stage W (transposed, f16) + bias into LDS ----
    for (int idx = tid; idx < DIN * DOUT; idx += NTHREADS) {
        int k = idx >> 6;      // input feature (row of W)
        int o = idx & 63;      // output column
        sW[o * LDK + k] = (_Float16)W[idx];
    }
    if (tid < DOUT) sBias[tid] = bias[tid];
    __syncthreads();

    for (int st = (int)blockIdx.x; st < n_super; st += (int)gridDim.x) {
        const int base = st * ROWS;   // flat edge index of first row

        // ---- Pass A: edata (contiguous 64x64 f32 block) -> sAct[:, 0:64) ----
        #pragma unroll
        for (int it = 0; it < 8; ++it) {
            int idx = (tid + it * NTHREADS) * 4;     // 0..4095
            int row = idx >> 6;
            int f0  = idx & 63;
            int g   = base + row;
            if (g < total) {
                v4f x = *(const v4f*)(edata + (size_t)g * DE + f0);
                v4h h = { (_Float16)x[0], (_Float16)x[1],
                          (_Float16)x[2], (_Float16)x[3] };
                *(v4h*)&sAct[row * LDK + f0] = h;
            }
        }

        // ---- Pass B: sender/receiver row gathers -> sAct[:, 64:192) ----
        {
            int row   = tid >> 1;          // 0..63
            int which = tid & 1;           // 0=sender, 1=receiver
            int g = base + row;
            if (g >= total) g = total - 1;
            int b   = g / E_PER_B;
            int vid = which ? receiver[g] : sender[g];
            const float* src = vdata + ((size_t)b * NVERT + (size_t)vid) * DV;
            _Float16* dst = &sAct[row * LDK + 64 + which * 64];
            #pragma unroll
            for (int j = 0; j < 16; ++j) {           // 64 floats = 256B row stream
                v4f x = *(const v4f*)(src + j * 4);
                v4h h = { (_Float16)x[0], (_Float16)x[1],
                          (_Float16)x[2], (_Float16)x[3] };
                *(v4h*)&dst[j * 4] = h;
            }
        }

        // ---- Pass C: cdata broadcast -> sAct[:, 192:224) ----
        #pragma unroll
        for (int it = 0; it < 4; ++it) {
            int idx = (tid + it * NTHREADS) * 4;     // 0..2047
            int row = idx >> 5;
            int f0  = idx & 31;
            int g = base + row;
            if (g >= total) g = total - 1;
            int b = g / E_PER_B;
            v4f x = *(const v4f*)(cdata + b * DC + f0);
            v4h h = { (_Float16)x[0], (_Float16)x[1],
                      (_Float16)x[2], (_Float16)x[3] };
            *(v4h*)&sAct[row * LDK + 192 + f0] = h;
        }

        __syncthreads();

        // ---- compute: each wave -> 16 edges x 64 outputs, K = 224 ----
        v8f acc[4] = {};                 // 4 N-tiles of 16
        const int arow  = lane & 15;     // A: M index
        const int ahalf = lane >> 4;     // A: K half selector (K+0..7 vs K+8..15)
        const _Float16* actRow = &sAct[(wave * 16 + arow) * LDK];
        const int bcol  = lane & 15;     // B: N index within tile
        const int bkoff = (lane >> 4) * 8;   // B: K half selector (K+0..7 vs K+8..15)

        #pragma unroll
        for (int k0 = 0; k0 < DIN; k0 += 32) {
            union { v16h v; v8h h[2]; } A;
            A.h[0] = *(const v8h*)&actRow[k0 + ahalf * 8];        // K = k0+{0..7|8..15}
            A.h[1] = *(const v8h*)&actRow[k0 + 16 + ahalf * 8];   // K = k0+16+{...}
            #pragma unroll
            for (int nt = 0; nt < 4; ++nt) {
                union { v16h v; v8h h[2]; } Bf;
                const _Float16* wp = &sW[(nt * 16 + bcol) * LDK + k0 + bkoff];
                Bf.h[0] = *(const v8h*)&wp[0];                    // K = k0+{0..7|8..15}
                Bf.h[1] = *(const v8h*)&wp[16];
                acc[nt] = __builtin_amdgcn_wmma_f32_16x16x32_f16(
                    false, A.v, false, Bf.v, (short)0, acc[nt], false, false);
            }
        }

        // ---- epilogue: bias + ReLU, scattered-row stores ----
        const int mbase = wave * 16 + (lane >> 4) * 8;  // C: VGPR r -> M = mbase + r
        #pragma unroll
        for (int nt = 0; nt < 4; ++nt) {
            int n = nt * 16 + bcol;
            float bv = sBias[n];
            #pragma unroll
            for (int r = 0; r < 8; ++r) {
                int g = base + mbase + r;
                if (g < total) {
                    float v = acc[nt][r] + bv;
                    out[(size_t)g * DOUT + n] = v > 0.0f ? v : 0.0f;
                }
            }
        }
        __syncthreads();   // before next iteration's act restaging
    }
}

extern "C" void kernel_launch(void* const* d_in, const int* in_sizes, int n_in,
                              void* d_out, int out_size, void* d_ws, size_t ws_size,
                              hipStream_t stream) {
    const float* edata    = (const float*)d_in[0];
    const float* vdata    = (const float*)d_in[1];
    const float* cdata    = (const float*)d_in[2];
    const int*   sender   = (const int*)d_in[3];
    const int*   receiver = (const int*)d_in[4];
    const float* W        = (const float*)d_in[5];
    const float* bias     = (const float*)d_in[6];
    float*       out      = (float*)d_out;

    int total   = in_sizes[3];                 // B*E flat edge count (800000)
    int n_super = (total + ROWS - 1) / ROWS;   // 12500
    int grid    = n_super < 2048 ? n_super : 2048;

    edgeblock_wmma<<<grid, NTHREADS, 0, stream>>>(
        edata, vdata, cdata, sender, receiver, W, bias, out, total, n_super);
}
